// SelfAttention_Realformer_86844238725642
// MI455X (gfx1250) — matrix operand order repaired and checked by reading, not compiled, not run on hardware
//
#include <hip/hip_runtime.h>
#include <hip/hip_bf16.h>

// Problem constants
#define B_   4
#define L_   1024
#define C_   768
#define H_   12
#define DK_  64
#define HDK_ 768
#define AST  1032   // padded LDS row stride (halves): 1032*2/4=516 dwords, 516%64=4 -> 16 rows hit distinct 4-bank groups

typedef __attribute__((ext_vector_type(16))) _Float16 v16h;
typedef __attribute__((ext_vector_type(8)))  _Float16 v8h;
typedef __attribute__((ext_vector_type(4)))  _Float16 v4h;
typedef __attribute__((ext_vector_type(8)))  float    v8f;

__device__ __forceinline__ v8f wmma_f16(v16h a, v16h b, v8f c) {
  // D = A(16x32 f16) * B(32x16 f16) + C(16x16 f32)
  return __builtin_amdgcn_wmma_f32_16x16x32_f16(false, a, false, b, (short)0, c,
                                                false, false);
}

// A-fragment from f32 row-major activations. Lane holds row m=lane&15;
// halves 0..7 = K[koff..koff+7], halves 8..15 = K[8+koff..8+koff+7].
__device__ __forceinline__ v16h load_a_f32(const float* __restrict__ p, int koff) {
  union { v16h v; _Float16 h[16]; } u;
  float4 x0 = *(const float4*)(p + koff);
  float4 x1 = *(const float4*)(p + koff + 4);
  float4 y0 = *(const float4*)(p + 8 + koff);
  float4 y1 = *(const float4*)(p + 8 + koff + 4);
  u.h[0]=(_Float16)x0.x;  u.h[1]=(_Float16)x0.y;  u.h[2]=(_Float16)x0.z;  u.h[3]=(_Float16)x0.w;
  u.h[4]=(_Float16)x1.x;  u.h[5]=(_Float16)x1.y;  u.h[6]=(_Float16)x1.z;  u.h[7]=(_Float16)x1.w;
  u.h[8]=(_Float16)y0.x;  u.h[9]=(_Float16)y0.y;  u.h[10]=(_Float16)y0.z; u.h[11]=(_Float16)y0.w;
  u.h[12]=(_Float16)y1.x; u.h[13]=(_Float16)y1.y; u.h[14]=(_Float16)y1.z; u.h[15]=(_Float16)y1.w;
  return u.v;
}

__device__ __forceinline__ v16h load_a_f16(const _Float16* __restrict__ p, int koff) {
  union { v16h v; v8h p2[2]; } u;
  u.p2[0] = *(const v8h*)(p + koff);        // 16B
  u.p2[1] = *(const v8h*)(p + 8 + koff);   // 16B
  return u.v;
}

// ---------------------------------------------------------------------------
// Weight transpose f32[K][N] -> f16[N][K] (coalesced via LDS tile)
// ---------------------------------------------------------------------------
__global__ void wtrans_kernel(const float* __restrict__ in, _Float16* __restrict__ out) {
  __shared__ float tile[32][33];
  int bx = blockIdx.x * 32, by = blockIdx.y * 32;
  int tx = threadIdx.x, ty = threadIdx.y;           // (32,8)
  #pragma unroll
  for (int i = 0; i < 32; i += 8)
    tile[ty + i][tx] = in[(size_t)(by + ty + i) * HDK_ + bx + tx];
  __syncthreads();
  #pragma unroll
  for (int i = 0; i < 32; i += 8)
    out[(size_t)(bx + ty + i) * C_ + by + tx] = (_Float16)tile[tx][ty + i];
}

// ---------------------------------------------------------------------------
// QKV projection: X[4096,768](f32) x Wt[N][K](f16) + b  -> q/k f16 [bh][L][DK],
// v f16 transposed [bh][DK][L].   block=256 (8 waves, wave=one 16x16 N-tile)
// ---------------------------------------------------------------------------
__global__ __launch_bounds__(256) void proj_qkv_kernel(
    const float* __restrict__ qx, const float* __restrict__ kx, const float* __restrict__ vx,
    const _Float16* __restrict__ wtq, const _Float16* __restrict__ wtk, const _Float16* __restrict__ wtv,
    const float* __restrict__ bq, const float* __restrict__ bk, const float* __restrict__ bv,
    _Float16* __restrict__ q, _Float16* __restrict__ k, _Float16* __restrict__ vt) {
  int mode = blockIdx.z;
  const float* A; const _Float16* Wt; const float* bias;
  if (mode == 0)      { A = qx; Wt = wtq; bias = bq; }
  else if (mode == 1) { A = kx; Wt = wtk; bias = bk; }
  else                { A = vx; Wt = wtv; bias = bv; }

  int t = threadIdx.x, lane = t & 31, w = t >> 5;
  int m0 = blockIdx.x * 16;
  int n0 = (blockIdx.y * 8 + w) * 16;
  int mrow = lane & 15, nl = lane & 15;
  int koff = (lane >> 4) * 16;
  int kbase = (lane >> 4) * 16;

  const float*    arow = A  + (size_t)(m0 + mrow) * C_;
  const _Float16* brow = Wt + (size_t)(n0 + nl) * C_;

  v8f acc = {0.f,0.f,0.f,0.f,0.f,0.f,0.f,0.f};
  for (int kb = 0; kb < C_; kb += 32) {
    v16h a = load_a_f32(arow + kb, koff);
    v16h b = *(const v16h*)(brow + kb + kbase);   // 32B contiguous
    acc = wmma_f16(a, b, acc);
  }

  float bn = bias[n0 + nl];
  int mb = (lane >> 4) * 8;
  int n = n0 + nl, h = n >> 6, d = n & 63;
  #pragma unroll
  for (int vg = 0; vg < 8; ++vg) {
    int row = m0 + mb + vg;
    int bb = row >> 10, l = row & (L_ - 1);
    float val = acc[vg] + bn;
    if (mode == 2)
      vt[(((size_t)bb * H_ + h) * DK_ + d) * L_ + l] = (_Float16)val;
    else {
      _Float16* dst = (mode == 0) ? q : k;
      dst[(((size_t)bb * H_ + h) * L_ + l) * DK_ + d] = (_Float16)val;
    }
  }
}

// ---------------------------------------------------------------------------
// Fused scores + softmax + alpha*V. One block = one (bh, 16-row strip).
// block=128 (4 waves).
//  Phase A: wave w computes 16 j-tiles of QK^T (WMMA), applies rel-pos/pre/mask,
//           streams f32 scores to d_out (non-temporal), keeps f16 copy in LDS,
//           tracks per-lane row maxima in registers.
//  Phase B: LDS max/sum reductions, exp in place (unnormalized, f16).
//  Phase C: alpha[16x1024] x V[1024x64] via WMMA (A from LDS, B=V^T contiguous),
//           1/rowsum folded into epilogue; writes zcat f16.
// ---------------------------------------------------------------------------
__global__ __launch_bounds__(128) void fused_attn_kernel(
    const _Float16* __restrict__ q, const _Float16* __restrict__ k,
    const _Float16* __restrict__ vt,
    const float* __restrict__ embK, const float* __restrict__ embB,
    const float* __restrict__ preS, const int* __restrict__ mask,
    float* __restrict__ sout, _Float16* __restrict__ zcat) {
  __shared__ alignas(32) _Float16 salpha[16 * AST];   // 33KB
  __shared__ float pmax[4 * 32 * 8];                  // 4KB per-lane row maxima
  __shared__ float sredm[128];
  __shared__ float srmax[16];
  __shared__ float srsum[16];
  __shared__ float sek[16], seb[16];

  int bh = blockIdx.y;
  int bb = bh / H_, h = bh - bb * H_;
  int i0 = blockIdx.x * 16;
  int t = threadIdx.x, lane = t & 31, w = t >> 5;
  if (t < 15) { sek[t] = embK[t * H_ + h]; seb[t] = embB[t * H_ + h]; }
  __syncthreads();

  int mrow = lane & 15, nl = lane & 15;
  int koff = (lane >> 4) * 16, kbase = (lane >> 4) * 16;
  int mb = (lane >> 4) * 8;

  // ---- Phase A: scores ----
  // Q fragments are invariant over j-tiles: hoist.
  const _Float16* qrow = q + ((size_t)bh * L_ + i0 + mrow) * DK_;
  v16h aq0 = load_a_f16(qrow, koff);
  v16h aq1 = load_a_f16(qrow + 32, koff);

  float vmax[8];
  #pragma unroll
  for (int vg = 0; vg < 8; ++vg) vmax[vg] = -3.4e38f;

  const _Float16* kmat = k + (size_t)bh * L_ * DK_;
  for (int iter = 0; iter < 16; ++iter) {
    int jt = iter * 4 + w;          // this wave's j-tile (0..63)
    int j0 = jt * 16;
    const _Float16* krow = kmat + (size_t)(j0 + nl) * DK_;
    v16h b0 = *(const v16h*)(krow + kbase);
    v16h b1 = *(const v16h*)(krow + 32 + kbase);
    v8f acc = {0.f,0.f,0.f,0.f,0.f,0.f,0.f,0.f};
    acc = wmma_f16(aq0, b0, acc);
    acc = wmma_f16(aq1, b1, acc);

    int j = j0 + nl;
    #pragma unroll
    for (int vg = 0; vg < 8; ++vg) {
      int i = i0 + mb + vg;
      int di = (i > j) ? (i - j) : (j - i);
      int idx = di;
      if (di > 7) {
        idx = 7 + (int)__log2f((float)(di - 7));
        if (idx > 14) idx = 14;
      }
      float s = acc[vg] * 0.125f * sek[idx] + seb[idx];
      size_t off = ((size_t)bh * L_ + i) * L_ + j;
      s += __builtin_nontemporal_load(preS + off);       // streamed once: NT
      if (mask[((size_t)bb * L_ + i) * L_ + j] == 0) s = -32768.0f;
      __builtin_nontemporal_store(s, sout + off);        // streamed once: NT
      vmax[vg] = fmaxf(vmax[vg], s);
      salpha[(mb + vg) * AST + j] = (_Float16)s;
    }
  }
  #pragma unroll
  for (int vg = 0; vg < 8; ++vg) pmax[(w * 32 + lane) * 8 + vg] = vmax[vg];
  __syncthreads();

  // ---- Phase B: row max, exp, row sum ----
  if (t < 16) {  // row t: contributions from lanes (t>>3)*16..+15 of each wave
    int hi = t >> 3, vg = t & 7;
    float m = -3.4e38f;
    #pragma unroll
    for (int ww = 0; ww < 4; ++ww)
      for (int n = 0; n < 16; ++n)
        m = fmaxf(m, pmax[(ww * 32 + hi * 16 + n) * 8 + vg]);
    srmax[t] = m;
  }
  __syncthreads();

  int r = t >> 3, sub = t & 7;
  float rm = srmax[r];
  float sum = 0.f;
  for (int it = 0; it < 32; ++it) {
    int c = sub * 4 + it * 32;
    #pragma unroll
    for (int jj = 0; jj < 4; ++jj) {
      float x = (float)salpha[r * AST + c + jj];
      float e = __expf(x - rm);
      salpha[r * AST + c + jj] = (_Float16)e;
      sum += e;
    }
  }
  sredm[t] = sum;
  __syncthreads();
  if (t < 16) {
    float s = 0.f;
    #pragma unroll
    for (int qq = 0; qq < 8; ++qq) s += sredm[t * 8 + qq];
    srsum[t] = s;
  }
  __syncthreads();

  // ---- Phase C: alpha x V ----
  int d = w * 16 + nl;
  const _Float16* vrow = vt + ((size_t)bh * DK_ + d) * L_;  // V^T: contiguous in j

  v8f acc = {0.f,0.f,0.f,0.f,0.f,0.f,0.f,0.f};
  for (int jt = 0; jt < 32; ++jt) {
    union { v16h v; v8h p2[2]; } a;
    a.p2[0] = *(const v8h*)&salpha[mrow * AST + jt * 32 + koff];
    a.p2[1] = *(const v8h*)&salpha[mrow * AST + jt * 32 + 8 + koff];
    v16h b = *(const v16h*)(vrow + jt * 32 + kbase);
    acc = wmma_f16(a.v, b, acc);
  }

  #pragma unroll
  for (int vg = 0; vg < 8; ++vg) {
    int mr = mb + vg;
    float val = acc[vg] / srsum[mr];
    int l = i0 + mr;
    zcat[((size_t)bb * L_ + l) * HDK_ + h * DK_ + d] = (_Float16)val;
  }
}

// ---------------------------------------------------------------------------
// Output projection: zcat[4096,768](f16) x WOt[N][K](f16) + b -> z f32 (d_out)
// ---------------------------------------------------------------------------
__global__ __launch_bounds__(256) void proj_out_kernel(
    const _Float16* __restrict__ zcat, const _Float16* __restrict__ wto,
    const float* __restrict__ bo, float* __restrict__ zout) {
  int t = threadIdx.x, lane = t & 31, w = t >> 5;
  int m0 = blockIdx.x * 16;
  int n0 = (blockIdx.y * 8 + w) * 16;
  int mrow = lane & 15, nl = lane & 15;
  int koff = (lane >> 4) * 16, kbase = (lane >> 4) * 16;

  const _Float16* arow = zcat + (size_t)(m0 + mrow) * HDK_;
  const _Float16* brow = wto + (size_t)(n0 + nl) * HDK_;

  v8f acc = {0.f,0.f,0.f,0.f,0.f,0.f,0.f,0.f};
  for (int kb = 0; kb < HDK_; kb += 32) {
    v16h a = load_a_f16(arow + kb, koff);
    v16h b = *(const v16h*)(brow + kb + kbase);
    acc = wmma_f16(a, b, acc);
  }

  float bn = bo[n0 + nl];
  int mb = (lane >> 4) * 8;
  #pragma unroll
  for (int vg = 0; vg < 8; ++vg) {
    int row = m0 + mb + vg;
    zout[(size_t)row * C_ + n0 + nl] = acc[vg] + bn;
  }
}

// ---------------------------------------------------------------------------
extern "C" void kernel_launch(void* const* d_in, const int* in_sizes, int n_in,
                              void* d_out, int out_size, void* d_ws, size_t ws_size,
                              hipStream_t stream) {
  const float* qx   = (const float*)d_in[0];
  const float* kx   = (const float*)d_in[1];
  const float* vx   = (const float*)d_in[2];
  const float* WQ_w = (const float*)d_in[3];
  const float* WQ_b = (const float*)d_in[4];
  const float* WK_w = (const float*)d_in[5];
  const float* WK_b = (const float*)d_in[6];
  const float* WV_w = (const float*)d_in[7];
  const float* WV_b = (const float*)d_in[8];
  const float* WO_w = (const float*)d_in[9];
  const float* WO_b = (const float*)d_in[10];
  const float* embK = (const float*)d_in[11];
  const float* embB = (const float*)d_in[12];
  const float* preS = (const float*)d_in[13];
  const int*   mask = (const int*)d_in[14];

  float* z_out = (float*)d_out;
  float* scores_out = z_out + (size_t)B_ * L_ * C_;   // second tuple element

  size_t off = 0;
  auto alloc = [&](size_t bytes) -> void* {
    void* p = (char*)d_ws + off;
    off += (bytes + 255) & ~(size_t)255;
    return p;
  };
  _Float16* wtq  = (_Float16*)alloc((size_t)HDK_ * C_ * 2);
  _Float16* wtk  = (_Float16*)alloc((size_t)HDK_ * C_ * 2);
  _Float16* wtv  = (_Float16*)alloc((size_t)HDK_ * C_ * 2);
  _Float16* wto  = (_Float16*)alloc((size_t)HDK_ * C_ * 2);
  _Float16* q    = (_Float16*)alloc((size_t)B_ * H_ * L_ * DK_ * 2);
  _Float16* k    = (_Float16*)alloc((size_t)B_ * H_ * L_ * DK_ * 2);
  _Float16* vt   = (_Float16*)alloc((size_t)B_ * H_ * DK_ * L_ * 2);
  _Float16* zcat = (_Float16*)alloc((size_t)B_ * L_ * HDK_ * 2);

  dim3 tb(32, 8), tg(HDK_ / 32, HDK_ / 32);
  wtrans_kernel<<<tg, tb, 0, stream>>>(WQ_w, wtq);
  wtrans_kernel<<<tg, tb, 0, stream>>>(WK_w, wtk);
  wtrans_kernel<<<tg, tb, 0, stream>>>(WV_w, wtv);
  wtrans_kernel<<<tg, tb, 0, stream>>>(WO_w, wto);

  proj_qkv_kernel<<<dim3((B_ * L_) / 16, HDK_ / 128, 3), 256, 0, stream>>>(
      qx, kx, vx, wtq, wtk, wtv, WQ_b, WK_b, WV_b, q, k, vt);

  fused_attn_kernel<<<dim3(L_ / 16, B_ * H_), 128, 0, stream>>>(
      q, k, vt, embK, embB, preS, mask, scores_out, zcat);

  proj_out_kernel<<<dim3((B_ * L_) / 16, C_ / 128), 256, 0, stream>>>(
      zcat, wto, WO_b, z_out);
}
